// WSM_RNN_67473936220688
// MI455X (gfx1250) — matrix operand order repaired and checked by reading, not compiled, not run on hardware
//
#include <hip/hip_runtime.h>
#include <hip/hip_bf16.h>
#include <math.h>

// ---------------------------------------------------------------------------
// WSM RNN scan for MI455X (gfx1250).
//
// thet_next = A@thet + B@d  with A = I + E, E in bf16 (values ~1e-4 so bf16
// relative rounding is harmless; keeping I in the fp32 "+thet" path preserves
// the state exactly).  E (81 MB bf16) stays resident in the 192 MB L2 and is
// streamed once per step via V_WMMA_F32_16X16X32_BF16 (M=6368,K=6368,N=16,
// batch 8 padded to 16).  64 sequential steps -> host loop, 3 kernels/step.
//
// GEMM step: one 256-thread block (8 waves) per 16-row M-tile; K is padded to
// 6400 (200 tiles of 32) and split 8 ways -> every wave runs a CONSTANT
// 25-iteration loop (uniform bounds => scalar branch, pipelined loads),
// partial f32 accumulators reduced through LDS.
// ---------------------------------------------------------------------------

#define LATENT  6368
#define KPAD    6400            // K padded to a multiple of 32*KWAVES
#define KTILESP (KPAD / 32)     // 200
#define MTILES  (LATENT / 16)   // 398
#define NB      8               // batch
#define TSTEPS  64
#define NDATA   16
#define OUTD    32
#define KWAVES  8               // K-split waves per M-tile
#define WITERS  (KTILESP / KWAVES)  // 25 K-tiles per wave (compile-time)
// theta layout offsets (fp32 flat vector)
#define OFF_W0  0
#define OFF_B0  64
#define OFF_W1  128
#define OFF_B1  4224
#define OFF_W2  4288
#define OFF_B2  6336

typedef __bf16 bf16_t;
typedef __attribute__((ext_vector_type(8)))  __bf16 v8bf;
typedef __attribute__((ext_vector_type(16))) __bf16 v16bf;
typedef __attribute__((ext_vector_type(8)))  float  v8f;

// ---------------------------------------------------------------------------
// Build E = A - I in bf16, row-major [LATENT][KPAD]; padded cols are zero.
// ---------------------------------------------------------------------------
__global__ __launch_bounds__(256) void build_E(const float* __restrict__ A,
                                               bf16_t* __restrict__ E) {
  size_t idx = (size_t)blockIdx.x * 256 + threadIdx.x;
  size_t n2 = (size_t)LATENT * KPAD;
  if (idx >= n2) return;
  int m = (int)(idx / KPAD);
  int k = (int)(idx % KPAD);
  float v = (k < LATENT) ? (A[(size_t)m * LATENT + k] - (m == k ? 1.0f : 0.0f))
                         : 0.0f;
  E[idx] = (bf16_t)v;
}

// ---------------------------------------------------------------------------
// Init: Theta fp32 [m][8]; bf16-transposed Theta [n][KPAD] with the pad
// region (rows n>=8 and column tail m>=LATENT) zeroed in BOTH ping-pong
// buffers (fill/zero address sets are disjoint -> no race).  Also initial
// x_hat state and d_0 = [t0 + t1, 0...].
// ---------------------------------------------------------------------------
__global__ __launch_bounds__(256) void init_ws(const float* __restrict__ xs,
                                               const float* __restrict__ ts,
                                               const float* __restrict__ th0,
                                               float*  __restrict__ T0,
                                               bf16_t* __restrict__ TT0,
                                               bf16_t* __restrict__ TT1,
                                               float*  __restrict__ dvec,
                                               float*  __restrict__ xhat) {
  int idx = blockIdx.x * 256 + threadIdx.x;
  if (idx < 16 * KPAD) {      // zero only the never-rewritten pad region
    int n = idx / KPAD;
    int m = idx % KPAD;
    if (n >= NB || m >= LATENT) {
      TT0[idx] = (bf16_t)0.0f;
      TT1[idx] = (bf16_t)0.0f;
    }
  }
  if (idx < LATENT * NB) {
    int m = idx >> 3, b = idx & 7;
    float v = th0[m];
    T0[idx] = v;
    TT0[(size_t)b * KPAD + m] = (bf16_t)v;
  }
  if (idx < NB * 17) {
    int b = idx / 17, j = idx % 17;
    dvec[idx] = (j == 0) ? (ts[b * TSTEPS] + ts[b * TSTEPS + 1]) : 0.0f;
  }
  if (idx < NB * NDATA) {
    int b = idx / NDATA, o = idx % NDATA;
    xhat[idx] = xs[((size_t)b * TSTEPS) * NDATA + o];
  }
}

// ---------------------------------------------------------------------------
// Bd[m][b] = sum_j B[m][j] * d[b][j]   (6368x17 matvec, 8 batches)
// ---------------------------------------------------------------------------
__global__ __launch_bounds__(256) void bd_step(const float* __restrict__ Bm,
                                               const float* __restrict__ dvec,
                                               float* __restrict__ Bd) {
  int idx = blockIdx.x * 256 + threadIdx.x;
  if (idx >= LATENT * NB) return;
  int m = idx >> 3, b = idx & 7;
  const float* br = Bm + (size_t)m * 17;
  const float* dv = dvec + b * 17;
  float s = 0.0f;
#pragma unroll
  for (int j = 0; j < 17; ++j) s = fmaf(br[j], dv[j], s);
  Bd[idx] = s;
}

// ---------------------------------------------------------------------------
// One scan step of the state GEMM:
//   Tdst[m][b] = Tsrc[m][b] + (E @ Tsrc_bf16)[m][b] + Bd[m][b]
// One block (8 waves) per 16-row M-tile; wave w handles K-tiles w, w+8, ...
// (exactly WITERS each, constant bounds).  Two interleaved accumulators break
// the WMMA->WMMA RAW chain; LDS reduction finishes one element per thread.
// A-operand (ISA 7.12.2, 16-bit A 16x32): lane r=l&15 is row m0+r,
//   halves c=l>>4 hold K [c*8, c*8+8) and [16+c*8, 24+c*8).
// B-operand: ThetaT bf16 column-major [n][KPAD] => lane n=l&15 reads
//   K [c*8, c*8+8) and [16+c*8, 24+c*8) of column n (pad region is zero).
// ---------------------------------------------------------------------------
__global__ __launch_bounds__(256) void gemm_step(const bf16_t* __restrict__ E,
                                                 const float*  __restrict__ Tsrc,
                                                 const bf16_t* __restrict__ TTsrc,
                                                 const float*  __restrict__ Bd,
                                                 float*  __restrict__ Tdst,
                                                 bf16_t* __restrict__ TTdst) {
  __shared__ float red[KWAVES][8][32];

  int wave = threadIdx.x >> 5;
  int lane = threadIdx.x & 31;
  int m0 = blockIdx.x * 16;          // grid.x == MTILES exactly
  int r  = lane & 15;
  int c  = lane >> 4;

  // per-lane base pointers for this wave's first K-tile (kt = wave)
  const bf16_t* ap = E + (size_t)(m0 + r) * KPAD + wave * 32;
  const bf16_t* bp = TTsrc + (size_t)r * KPAD + wave * 32;

  v8f acc0 = {0.f, 0.f, 0.f, 0.f, 0.f, 0.f, 0.f, 0.f};
  v8f acc1 = acc0;

  auto body = [&](int i, v8f& acc) {
    const bf16_t* a = ap + (size_t)i * (KWAVES * 32);
    const bf16_t* b = bp + (size_t)i * (KWAVES * 32);
    v8bf a0 = *(const v8bf*)(a + c * 8);
    v8bf a1 = *(const v8bf*)(a + 16 + c * 8);
    v16bf av = __builtin_shufflevector(a0, a1, 0, 1, 2, 3, 4, 5, 6, 7,
                                       8, 9, 10, 11, 12, 13, 14, 15);
    v8bf b0 = *(const v8bf*)(b + c * 8);
    v8bf b1 = *(const v8bf*)(b + 16 + c * 8);
    v16bf bv = __builtin_shufflevector(b0, b1, 0, 1, 2, 3, 4, 5, 6, 7,
                                       8, 9, 10, 11, 12, 13, 14, 15);
    acc = __builtin_amdgcn_wmma_f32_16x16x32_bf16(
        false, av, false, bv, (short)0, acc, false, false);
  };

  // WITERS = 25: 12 constant-bound double iterations + static tail
  for (int i = 0; i < WITERS - 1; i += 2) {
    body(i, acc0);
    body(i + 1, acc1);
  }
  body(WITERS - 1, acc0);

  v8f acc = acc0 + acc1;

  // park partial sums in LDS: red[wave][rr][lane]
#pragma unroll
  for (int rr = 0; rr < 8; ++rr) red[wave][rr][lane] = acc[rr];
  __syncthreads();

  // 256 threads == 8 rr x 32 lanes: thread (rr = tid>>5, lane) reduces and
  // finishes one output element.
  int rr = wave;                      // reuse tid>>5 as the VGPR-row index
  float s = 0.0f;
#pragma unroll
  for (int w = 0; w < KWAVES; ++w) s += red[w][rr][lane];

  // C/D layout: (rr, lane) -> M = m0 + rr + (lane>>4)*8, N = lane & 15.
  int n = lane & 15;
  int m = m0 + rr + (lane >> 4) * 8;
  if (n < NB) {
    float v = Tsrc[(size_t)m * NB + n] + s + Bd[(size_t)m * NB + n];
    Tdst[(size_t)m * NB + n] = v;
    TTdst[(size_t)n * KPAD + m] = (bf16_t)v;   // bf16 transposed copy
  }
}

// ---------------------------------------------------------------------------
// MLP head + output + state/delta update.  One wave per batch element.
// ---------------------------------------------------------------------------
__global__ __launch_bounds__(256) void mlp_step(const float* __restrict__ theta,
                                                const float* __restrict__ xs,
                                                const float* __restrict__ ts,
                                                float* __restrict__ out,
                                                float* __restrict__ dvec,
                                                float* __restrict__ xhat,
                                                int t) {
  __shared__ float h0[NB][64];
  __shared__ float h1[NB][64];
  int b = threadIdx.x >> 5;
  int lane = threadIdx.x & 31;

  float tc = ts[b * TSTEPS + t];
  float tp = (t == 0) ? -ts[b * TSTEPS + 1] : ts[b * TSTEPS + t - 1];
  float root = 2.0f * tc - tp;  // t_curr + delta_t

  auto TH = [&](int m) { return theta[(size_t)m * NB + b]; };

  // layer 0: h0 = relu(W0 * root + b0)
  for (int j = lane; j < 64; j += 32) {
    float v = fmaf(TH(OFF_W0 + j), root, TH(OFF_B0 + j));
    h0[b][j] = v > 0.0f ? v : 0.0f;
  }
  __syncthreads();

  // layer 1: h1 = relu(W1 @ h0 + b1)
  for (int j = lane; j < 64; j += 32) {
    float s = TH(OFF_B1 + j);
    for (int i = 0; i < 64; ++i) s = fmaf(TH(OFF_W1 + j * 64 + i), h0[b][i], s);
    h1[b][j] = s > 0.0f ? s : 0.0f;
  }
  __syncthreads();

  // layer 2: out = W2 @ h1 + b2 ; heads
  float s = TH(OFF_B2 + lane);
  for (int i = 0; i < 64; ++i) s = fmaf(TH(OFF_W2 + lane * 64 + i), h1[b][i], s);
  float th = tanhf(s);
  float y = (lane < NDATA) ? th : log1pf(expf(th));   // mean | softplus(tanh)
  out[((size_t)b * TSTEPS + t) * OUTD + lane] = y;

  // d_{t+1} = [t_{t+1} - t_t, xs[t+1] - x_hat_t]  (x_hat BEFORE update);
  // then x_hat <- mean_t.
  if (lane < NDATA) {
    float xh = xhat[b * NDATA + lane];
    if (t < TSTEPS - 1) {
      float xn = xs[((size_t)b * TSTEPS + t + 1) * NDATA + lane];
      dvec[b * 17 + 1 + lane] = xn - xh;
      if (lane == 0) dvec[b * 17] = ts[b * TSTEPS + t + 1] - tc;
    }
    xhat[b * NDATA + lane] = y;  // y == mean for lane < 16
  }
}

// ---------------------------------------------------------------------------
extern "C" void kernel_launch(void* const* d_in, const int* in_sizes, int n_in,
                              void* d_out, int out_size, void* d_ws, size_t ws_size,
                              hipStream_t stream) {
  (void)in_sizes; (void)n_in; (void)out_size; (void)ws_size;
  const float* xs  = (const float*)d_in[0];   // (8,64,16)
  const float* ts  = (const float*)d_in[1];   // (8,64)
  const float* A   = (const float*)d_in[2];   // (6368,6368)
  const float* Bm  = (const float*)d_in[3];   // (6368,17)
  const float* th0 = (const float*)d_in[4];   // (6368,)
  float* out = (float*)d_out;                 // (8,64,32)

  // workspace carve-out (~82 MB total)
  char* ws = (char*)d_ws;
  size_t off = 0;
  auto carve = [&](size_t bytes) -> void* {
    void* p = ws + off;
    off += (bytes + 255) & ~(size_t)255;
    return p;
  };
  const size_t n2 = (size_t)LATENT * KPAD;
  bf16_t* E    = (bf16_t*)carve(n2 * sizeof(bf16_t));
  float*  Tf0  = (float*) carve((size_t)LATENT * NB * sizeof(float));
  float*  Tf1  = (float*) carve((size_t)LATENT * NB * sizeof(float));
  bf16_t* TT0  = (bf16_t*)carve((size_t)16 * KPAD * sizeof(bf16_t));
  bf16_t* TT1  = (bf16_t*)carve((size_t)16 * KPAD * sizeof(bf16_t));
  float*  Bd   = (float*) carve((size_t)LATENT * NB * sizeof(float));
  float*  dvec = (float*) carve((size_t)NB * 17 * sizeof(float));
  float*  xhat = (float*) carve((size_t)NB * NDATA * sizeof(float));

  int nconv = (int)((n2 + 255) / 256);
  build_E<<<nconv, 256, 0, stream>>>(A, E);
  init_ws<<<(16 * KPAD + 255) / 256, 256, 0, stream>>>(xs, ts, th0, Tf0, TT0,
                                                       TT1, dvec, xhat);

  for (int t = 0; t < TSTEPS; ++t) {
    float*  Ts  = (t & 1) ? Tf1 : Tf0;
    float*  Td  = (t & 1) ? Tf0 : Tf1;
    bf16_t* TTs = (t & 1) ? TT1 : TT0;
    bf16_t* TTd = (t & 1) ? TT0 : TT1;

    bd_step<<<(LATENT * NB + 255) / 256, 256, 0, stream>>>(Bm, dvec, Bd);
    gemm_step<<<MTILES, 256, 0, stream>>>(E, Ts, TTs, Bd, Td, TTd);
    mlp_step<<<1, 256, 0, stream>>>(Td, xs, ts, out, dvec, xhat, t);
  }
}
